// Head_3178275799155
// MI455X (gfx1250) — matrix operand order repaired and checked by reading, not compiled, not run on hardware
//
#include <hip/hip_runtime.h>

// ---------------------------------------------------------------------------
// Single-head causal attention for MI455X (gfx1250, wave32, WMMA).
//   B=4, S=4096, D=1024, H=64, fp32 in/out, bf16 WMMA internally (f32 accum).
// Kernel 1: fused QKV projection (reads x once: bandwidth-bound, ~64MB).
// Kernel 2: flash attention, double-buffered async K/V staging into LDS.
// ---------------------------------------------------------------------------

constexpr int Bn = 4, Sn = 4096, Dn = 1024, Hn = 64;
constexpr int Mn = Bn * Sn;                 // 16384 flattened rows

typedef __attribute__((ext_vector_type(16))) __bf16        v16bf;
typedef __attribute__((ext_vector_type(8)))  float         v8f;
typedef __attribute__((ext_vector_type(4)))  unsigned int  v4u;
typedef __attribute__((ext_vector_type(4)))  int           v4i;

#define AS1 __attribute__((address_space(1)))
#define AS3 __attribute__((address_space(3)))

#if __has_builtin(__builtin_amdgcn_global_load_async_to_lds_b128)
#define HAVE_ASYNC_LDS 1
#else
#define HAVE_ASYNC_LDS 0
#endif

struct FragU4  { v4u lo, hi; };             // 32B, two contiguous 16B runs
struct FragU16 { unsigned short v[16]; };   // 32B, element-wise build

static __device__ __forceinline__ v16bf frag_cast(FragU4 f) {
  return __builtin_bit_cast(v16bf, f);
}
static __device__ __forceinline__ v16bf frag_cast(FragU16 f) {
  return __builtin_bit_cast(v16bf, f);
}

// fp32 -> bf16 round-to-nearest-even
static __device__ __forceinline__ unsigned short f2bf(float f) {
  unsigned u = __float_as_uint(f);
  u += 0x7fffu + ((u >> 16) & 1u);
  return (unsigned short)(u >> 16);
}

static __device__ __forceinline__ v8f wmma_bf16(v16bf a, v16bf b, v8f c) {
  // D = A(16x32 bf16) * B(32x16 bf16) + C(16x16 f32)
  return __builtin_amdgcn_wmma_f32_16x16x32_bf16(false, a, false, b,
                                                 (short)0, c, false, false);
}

// Builtin signature (from clang): (int4 AS1*, int4 AS3*, imm offset, imm cpol)
static __device__ __forceinline__ AS1 v4i* gcast(const void* p) {
  return (AS1 v4i*)(unsigned long long)(size_t)p;
}
static __device__ __forceinline__ AS3 v4i* lcast(void* p) {
  // flat shared address: low 32 bits are the LDS byte offset
  return (AS3 v4i*)(unsigned)(size_t)p;
}

static __device__ __forceinline__ void wait_async_lds() {
#if HAVE_ASYNC_LDS
#if __has_builtin(__builtin_amdgcn_s_wait_asynccnt)
  __builtin_amdgcn_s_wait_asynccnt(0);
#else
  asm volatile("s_wait_asynccnt 0" ::: "memory");
#endif
#endif
}

// ---------------------------------------------------------------------------
// Kernel 1: Q/K/V projection.  grid = Mn/64 blocks, 128 threads (4 waves).
// Wave w owns rows [rowbase .. rowbase+15].  K staged in 32-deep chunks.
// ---------------------------------------------------------------------------
__global__ void __launch_bounds__(128)
qkv_proj_kernel(const float* __restrict__ x,
                const float* __restrict__ Wq,
                const float* __restrict__ Wk,
                const float* __restrict__ Wv,
                unsigned short* __restrict__ ws)   // [3][Mn][Hn] bf16
{
  __shared__ unsigned short lW[3][64 * 32];        // transposed: [n][k] bf16

  const int tid  = threadIdx.x;
  const int wave = tid >> 5;
  const int lane = tid & 31;
  const int r    = lane & 15;                      // row / col-within-tile
  const int hi   = lane >> 4;                      // lane half select
  const int rowbase = blockIdx.x * 64 + wave * 16;

  v8f acc[3][4];
#pragma unroll
  for (int p = 0; p < 3; ++p)
#pragma unroll
    for (int t = 0; t < 4; ++t)
      acc[p][t] = v8f{0.f, 0.f, 0.f, 0.f, 0.f, 0.f, 0.f, 0.f};

  for (int kc = 0; kc < Dn / 32; ++kc) {
    const int kbase = kc * 32;

    // Cooperative load of W chunks, fp32 -> bf16, transposed to [n][k].
    for (int idx = tid; idx < 3 * 2048; idx += 128) {
      const int p  = idx >> 11;
      const int r2 = idx & 2047;
      const int n  = r2 & 63;                      // consecutive -> coalesced
      const int k  = r2 >> 6;
      const float* Wp = (p == 0) ? Wq : (p == 1) ? Wk : Wv;
      lW[p][n * 32 + k] = f2bf(Wp[(size_t)(kbase + k) * Hn + n]);
    }
    __syncthreads();

    // A fragment: rows of x (fp32 -> bf16).
    // A 16x32 layout: lane row = r; K = hi*16 + g*8 + w  (g=0,1; w=0..7)
    FragU16 ax;
    {
      const float* xr = x + (size_t)(rowbase + r) * Dn + kbase;
      if (kc + 1 < Dn / 32) __builtin_prefetch(xr + 32, 0, 1);  // next chunk
      const int o0 = hi * 16, o1 = hi * 16 + 8;
#pragma unroll
      for (int w = 0; w < 8; ++w) ax.v[w]     = f2bf(xr[o0 + w]);
#pragma unroll
      for (int w = 0; w < 8; ++w) ax.v[8 + w] = f2bf(xr[o1 + w]);
    }
    const v16bf a = frag_cast(ax);

#pragma unroll
    for (int p = 0; p < 3; ++p) {
#pragma unroll
      for (int t = 0; t < 4; ++t) {
        // B 32x16 layout: lane col n = t*16+r; K = hi*16 + e (contiguous)
        const v4u* src = (const v4u*)&lW[p][(t * 16 + r) * 32 + hi * 16];
        FragU4 fb; fb.lo = src[0]; fb.hi = src[1];
        acc[p][t] = wmma_bf16(a, frag_cast(fb), acc[p][t]);
      }
    }
    __syncthreads();
  }

  // C layout: VGPR j -> row j + hi*8, col = t*16 + r.  Store bf16 to ws.
#pragma unroll
  for (int p = 0; p < 3; ++p)
#pragma unroll
    for (int t = 0; t < 4; ++t)
#pragma unroll
      for (int j = 0; j < 8; ++j) {
        const int row = rowbase + j + hi * 8;
        ws[(size_t)p * Mn * Hn + (size_t)row * Hn + t * 16 + r] =
            f2bf(acc[p][t][j]);
      }
}

// ---------------------------------------------------------------------------
// Kernel 2: flash attention.  grid = B*(S/64) blocks, 128 threads (4 waves).
// Wave w owns 16 query rows; block shares K/V 32-key chunks via LDS with
// double-buffered async (ASYNCcnt) staging.
// ---------------------------------------------------------------------------
__global__ void __launch_bounds__(128)
attn_kernel(const unsigned short* __restrict__ qws,
            const unsigned short* __restrict__ kws,
            const unsigned short* __restrict__ vws,
            float* __restrict__ out)
{
  __shared__ unsigned short lK[2][32 * 64];  // [buf][key][h] bf16
  __shared__ unsigned short lV[2][32 * 64];  // [buf][key][h] bf16
  __shared__ unsigned short lP[4][16 * 32];  // per-wave P transpose buffer

  const int tid  = threadIdx.x;
  const int wave = tid >> 5;
  const int lane = tid & 31;
  const int r    = lane & 15;
  const int hi   = lane >> 4;

  const int b         = blockIdx.x >> 6;     // S/64 = 64 q-blocks per batch
  const int qbase_blk = (blockIdx.x & 63) * 64;
  const int qbase     = qbase_blk + wave * 16;
  const int qend      = qbase + 15;

  // Stage one 32-key K/V chunk (4 KB each) into LDS buffer `buf`.
  auto stage = [&](int buf, int ch) {
    const unsigned short* gk = kws + ((size_t)b * Sn + ch * 32) * Hn;
    const unsigned short* gv = vws + ((size_t)b * Sn + ch * 32) * Hn;
#if HAVE_ASYNC_LDS
#pragma unroll
    for (int i = 0; i < 2; ++i) {
      const int idx = tid + i * 128;             // 256 x 16B per array
      __builtin_amdgcn_global_load_async_to_lds_b128(
          gcast(gk + idx * 8), lcast(&lK[buf][idx * 8]), 0, 0);
      __builtin_amdgcn_global_load_async_to_lds_b128(
          gcast(gv + idx * 8), lcast(&lV[buf][idx * 8]), 0, 0);
    }
#else
    const v4u* gks = (const v4u*)gk;
    const v4u* gvs = (const v4u*)gv;
    v4u* sk = (v4u*)lK[buf];
    v4u* sv = (v4u*)lV[buf];
#pragma unroll
    for (int i = 0; i < 2; ++i) {
      sk[tid + i * 128] = gks[tid + i * 128];
      sv[tid + i * 128] = gvs[tid + i * 128];
    }
#endif
  };

  // Q fragments for the two K=32 slices of H=64 (A layout), loaded once.
  v16bf aq[2];
#pragma unroll
  for (int c = 0; c < 2; ++c) {
    const unsigned short* qrow = qws + ((size_t)b * Sn + qbase + r) * Hn + c * 32;
    FragU4 f;
    f.lo = *(const v4u*)(qrow + hi * 16);
    f.hi = *(const v4u*)(qrow + hi * 16 + 8);
    aq[c] = frag_cast(f);
  }

  v8f o[4];
#pragma unroll
  for (int t = 0; t < 4; ++t)
    o[t] = v8f{0.f, 0.f, 0.f, 0.f, 0.f, 0.f, 0.f, 0.f};
  float m[8], l[8];
#pragma unroll
  for (int j = 0; j < 8; ++j) { m[j] = -INFINITY; l[j] = 0.f; }

  const int nchunks = (qbase_blk + 64) / 32;     // block-uniform
  stage(0, 0);

  for (int ch = 0; ch < nchunks; ++ch) {
    const int kbase = ch * 32;
    const int buf   = ch & 1;

    wait_async_lds();          // this wave's staging of `buf` is complete
    __syncthreads();           // everyone's staging of `buf` is complete

    if (ch + 1 < nchunks) stage(buf ^ 1, ch + 1);   // overlap with compute

    if (kbase <= qend) {                       // wave-uniform: EXEC stays full
      // ---- S = Q * K^T  (two 16-key tiles, K-dim = H in two 32 slices) ----
      v8f sc[2];
#pragma unroll
      for (int t2 = 0; t2 < 2; ++t2) {
        v8f cacc = v8f{0.f, 0.f, 0.f, 0.f, 0.f, 0.f, 0.f, 0.f};
#pragma unroll
        for (int c = 0; c < 2; ++c) {
          // B layout: col n = key t2*16+r; K = h = c*32 + hi*16 + e
          const v4u* src =
              (const v4u*)&lK[buf][(t2 * 16 + r) * 64 + c * 32 + hi * 16];
          FragU4 fb; fb.lo = src[0]; fb.hi = src[1];
          cacc = wmma_bf16(aq[c], frag_cast(fb), cacc);
        }
        sc[t2] = cacc;
      }

      // ---- scale + causal mask + row max ----
      float cmax[8];
#pragma unroll
      for (int j = 0; j < 8; ++j) {
        const int grow = qbase + j + hi * 8;
        cmax[j] = -INFINITY;
#pragma unroll
        for (int t2 = 0; t2 < 2; ++t2) {
          const int col = kbase + t2 * 16 + r;
          float s = sc[t2][j] * 0.125f;        // H^-0.5
          if (col > grow) s = -INFINITY;
          sc[t2][j] = s;
          cmax[j] = fmaxf(cmax[j], s);
        }
      }
#pragma unroll
      for (int j = 0; j < 8; ++j)
#pragma unroll
        for (int off = 1; off < 16; off <<= 1)
          cmax[j] = fmaxf(cmax[j], __shfl_xor(cmax[j], off, 32));

      float scal[8], psum[8];
#pragma unroll
      for (int j = 0; j < 8; ++j) {
        const float mn = fmaxf(m[j], cmax[j]);
        scal[j] = __expf(m[j] - mn);
        m[j] = mn;
        psum[j] = 0.f;
      }

      // ---- P = exp(S - m): accumulate row sums, stash bf16 in LDS ----
#pragma unroll
      for (int t2 = 0; t2 < 2; ++t2)
#pragma unroll
        for (int j = 0; j < 8; ++j) {
          const float p = __expf(sc[t2][j] - m[j]);
          psum[j] += p;
          lP[wave][(j + hi * 8) * 32 + t2 * 16 + r] = f2bf(p);
        }
#pragma unroll
      for (int j = 0; j < 8; ++j) {
#pragma unroll
        for (int off = 1; off < 16; off <<= 1)
          psum[j] += __shfl_xor(psum[j], off, 32);
        l[j] = l[j] * scal[j] + psum[j];
      }

      // rescale running O
#pragma unroll
      for (int t = 0; t < 4; ++t)
#pragma unroll
        for (int j = 0; j < 8; ++j)
          o[t][j] *= scal[j];

      // LDS is in-order per wave, but make the RAW explicit.
      asm volatile("s_wait_dscnt 0" ::: "memory");

      // ---- re-load P in A layout: row r, K = hi*16 + g*8 + w ----
      FragU4 fp;
      fp.lo = *(const v4u*)&lP[wave][r * 32 + hi * 16];
      fp.hi = *(const v4u*)&lP[wave][r * 32 + hi * 16 + 8];
      const v16bf ap = frag_cast(fp);

      // ---- O += P * V  (4 h-tiles; B element (k,n) = V[k][t*16+n]) ----
#pragma unroll
      for (int t = 0; t < 4; ++t) {
        FragU16 bv;
#pragma unroll
        for (int e = 0; e < 16; ++e)
          bv.v[e] = lV[buf][(hi * 16 + e) * 64 + t * 16 + r];
        o[t] = wmma_bf16(ap, frag_cast(bv), o[t]);
      }
    }
    __syncthreads();           // all waves done with `buf` before re-staging
  }

  // ---- epilogue: divide by row sum, store fp32 ----
#pragma unroll
  for (int t = 0; t < 4; ++t)
#pragma unroll
    for (int j = 0; j < 8; ++j) {
      const int row = qbase + j + hi * 8;
      out[((size_t)b * Sn + row) * Hn + t * 16 + r] = o[t][j] * (1.0f / l[j]);
    }
}

// ---------------------------------------------------------------------------
extern "C" void kernel_launch(void* const* d_in, const int* in_sizes, int n_in,
                              void* d_out, int out_size, void* d_ws,
                              size_t ws_size, hipStream_t stream) {
  (void)in_sizes; (void)n_in; (void)out_size; (void)ws_size;

  const float* x  = (const float*)d_in[0];
  const float* Wq = (const float*)d_in[1];
  const float* Wk = (const float*)d_in[2];
  const float* Wv = (const float*)d_in[3];

  unsigned short* ws  = (unsigned short*)d_ws;        // 3 * 2 MB bf16 QKV
  unsigned short* qws = ws;
  unsigned short* kws = ws + (size_t)1 * Mn * Hn;
  unsigned short* vws = ws + (size_t)2 * Mn * Hn;

  qkv_proj_kernel<<<dim3(Mn / 64), dim3(128), 0, stream>>>(x, Wq, Wk, Wv, ws);
  attn_kernel<<<dim3(Bn * (Sn / 64)), dim3(128), 0, stream>>>(
      qws, kws, vws, (float*)d_out);
}
